// SelfAttention_22677427323425
// MI455X (gfx1250) — matrix operand order repaired and checked by reading, compiled, not run on hardware
//
#include <hip/hip_runtime.h>
#include <hip/hip_bf16.h>
#include <math.h>

// ---------------- problem constants (from reference) ----------------
#define DIMD   1024
#define NSEQ   1024
#define BATCH  8
#define NHEAD  16
#define DKH    64
#define HID    4096
#define E3     3072
#define MROWS  (BATCH*NSEQ)   // 8192

// ---------------- WMMA types ----------------
typedef __attribute__((ext_vector_type(16))) __bf16 v16bf;
typedef __attribute__((ext_vector_type(8)))  float  v8f;

union FragBF { v16bf v; uint4 q[2]; };
union FragF  { v8f   v; float f[8]; };

__device__ __forceinline__ __bf16 f2bf(float f) { return (__bf16)f; }

__device__ __forceinline__ v8f wmma_bf16(const FragBF& a, const FragBF& b, v8f c) {
  // D = A(16x32 bf16) * B(32x16 bf16) + C(16x16 f32)
  return __builtin_amdgcn_wmma_f32_16x16x32_bf16(false, a.v, false, b.v,
                                                 (short)0, c, false, false);
}

// ---------------- gfx1250 async global->LDS DMA (ASYNCcnt path) ----------------
typedef __attribute__((address_space(3))) void lds_void;

__device__ __forceinline__ uint32_t lds_addr(void* p) {
  return (uint32_t)(uintptr_t)(lds_void*)p;   // generic -> LDS byte offset
}

__device__ __forceinline__ void async_b128(uint32_t lds, const void* g) {
  // VDST = LDS byte address, VADDR = 64-bit global address, no SADDR
  asm volatile("global_load_async_to_lds_b128 %0, %1, off"
               :: "v"(lds), "v"((unsigned long long)(uintptr_t)g) : "memory");
}

template<int N>
__device__ __forceinline__ void wait_asynccnt() {
  asm volatile("s_wait_asynccnt %0" :: "i"(N) : "memory");
}

// ---------------- f32 -> bf16 conversion ----------------
__global__ void cvt_bf16_kernel(const float* __restrict__ in,
                                __bf16* __restrict__ out, int n) {
  int i = blockIdx.x * blockDim.x + threadIdx.x;
  int stride = gridDim.x * blockDim.x;
  for (; i < n; i += stride) out[i] = f2bf(in[i]);
}

// ---------------- LayerNorm (one block per row of 1024) ----------------
__global__ __launch_bounds__(256)
void ln_bf16_kernel(const float* __restrict__ x, const float* __restrict__ g,
                    const float* __restrict__ be, __bf16* __restrict__ out) {
  __shared__ float s1[256];
  __shared__ float s2[256];
  const int row = blockIdx.x, t = threadIdx.x;
  const float* xr = x + (size_t)row * DIMD;
  float4 xv = ((const float4*)xr)[t];
  s1[t] = xv.x + xv.y + xv.z + xv.w;
  s2[t] = xv.x*xv.x + xv.y*xv.y + xv.z*xv.z + xv.w*xv.w;
  __syncthreads();
  for (int off = 128; off > 0; off >>= 1) {
    if (t < off) { s1[t] += s1[t+off]; s2[t] += s2[t+off]; }
    __syncthreads();
  }
  const float mu   = s1[0] * (1.0f / DIMD);
  const float var  = s2[0] * (1.0f / DIMD) - mu * mu;
  const float rstd = rsqrtf(var + 1e-6f);
  float4 gv = ((const float4*)g)[t];
  float4 bv = ((const float4*)be)[t];
  __bf16* orow = out + (size_t)row * DIMD + t * 4;
  orow[0] = f2bf((xv.x - mu) * rstd * gv.x + bv.x);
  orow[1] = f2bf((xv.y - mu) * rstd * gv.y + bv.y);
  orow[2] = f2bf((xv.z - mu) * rstd * gv.z + bv.z);
  orow[3] = f2bf((xv.w - mu) * rstd * gv.w + bv.w);
}

// ---------------- tiled bf16 GEMM: C[M,N] = A[M,K] * W[N,K]^T ----------------
// 128x128 block tile, 8 waves of 32x64, K-step 32, async double-buffered LDS.
// EPI 0: store bf16             (qkv)
// EPI 1: +bias +residual, f32   (proj / fc2)
// EPI 2: +bias, exact GELU, bf16 (fc1)
template<int EPI>
__global__ __launch_bounds__(256)
void gemm_bf16_kernel(const __bf16* __restrict__ A, const __bf16* __restrict__ W,
                      const float* __restrict__ bias, const float* __restrict__ res,
                      void* __restrict__ outp, int M, int N, int K) {
  __shared__ __align__(16) __bf16 As[2][128][40];  // +8 pad, 16B-aligned rows
  __shared__ __align__(16) __bf16 Bs[2][128][40];

  const int tid  = threadIdx.x;
  const int wave = tid >> 5, lane = tid & 31;
  const int hf   = lane >> 4, l16 = lane & 15;
  const int wm   = wave >> 1, wn  = wave & 1;     // 4x2 waves -> 32x64 tiles
  const int m0   = blockIdx.y * 128;
  const int n0   = blockIdx.x * 128;

  FragF acc[2][4];
  v8f zero = {};
#pragma unroll
  for (int mi = 0; mi < 2; ++mi)
#pragma unroll
    for (int ni = 0; ni < 4; ++ni) acc[mi][ni].v = zero;

  // staging: 256 threads move 128 rows x 32 K of A and of W (32B = 2 b128 each)
  const int sr = tid >> 1, sk = (tid & 1) * 16;

  const int ksteps = K >> 5;
  auto issue = [&](int kt, int buf) {
    const __bf16* ap = A + (size_t)(m0 + sr) * K + kt * 32 + sk;
    async_b128(lds_addr(&As[buf][sr][sk]),     ap);
    async_b128(lds_addr(&As[buf][sr][sk + 8]), ap + 8);
    const __bf16* wp = W + (size_t)(n0 + sr) * K + kt * 32 + sk;
    async_b128(lds_addr(&Bs[buf][sr][sk]),     wp);
    async_b128(lds_addr(&Bs[buf][sr][sk + 8]), wp + 8);
  };

  issue(0, 0);
  for (int kt = 0; kt < ksteps; ++kt) {
    const int cur = kt & 1;
    if (kt + 1 < ksteps) {
      issue(kt + 1, cur ^ 1);      // DMA next slice while computing this one
      wait_asynccnt<4>();          // 4 newer ops outstanding -> current slice landed
    } else {
      wait_asynccnt<0>();
    }
    __syncthreads();

    FragBF a[2], b[4];
#pragma unroll
    for (int mi = 0; mi < 2; ++mi) {   // A frag: half0 K0..7|16..23, half1 K8..15|24..31
      const int r = wm * 32 + mi * 16 + l16;
      a[mi].q[0] = *(const uint4*)&As[cur][r][hf * 8];
      a[mi].q[1] = *(const uint4*)&As[cur][r][16 + hf * 8];
    }
#pragma unroll
    for (int ni = 0; ni < 4; ++ni) {   // B frag: lane=col, half0 K0..7|16..23, half1 K8..15|24..31
      const int c = wn * 64 + ni * 16 + l16;
      b[ni].q[0] = *(const uint4*)&Bs[cur][c][hf * 8];
      b[ni].q[1] = *(const uint4*)&Bs[cur][c][16 + hf * 8];
    }
#pragma unroll
    for (int mi = 0; mi < 2; ++mi)
#pragma unroll
      for (int ni = 0; ni < 4; ++ni)
        acc[mi][ni].v = wmma_bf16(a[mi], b[ni], acc[mi][ni].v);
    __syncthreads();
  }

  // epilogue: C layout -> row = i + 8*hf, col = l16 (per 16x16 tile)
#pragma unroll
  for (int mi = 0; mi < 2; ++mi) {
#pragma unroll
    for (int ni = 0; ni < 4; ++ni) {
      const int col = n0 + wn * 64 + ni * 16 + l16;
#pragma unroll
      for (int i = 0; i < 8; ++i) {
        const int row = m0 + wm * 32 + mi * 16 + i + hf * 8;
        float v = acc[mi][ni].f[i];
        if (EPI == 0) {
          ((__bf16*)outp)[(size_t)row * N + col] = f2bf(v);
        } else if (EPI == 1) {
          v += bias[col] + res[(size_t)row * N + col];
          ((float*)outp)[(size_t)row * N + col] = v;
        } else {
          v += bias[col];
          v = 0.5f * v * (1.0f + erff(v * 0.70710678118654752f));
          ((__bf16*)outp)[(size_t)row * N + col] = f2bf(v);
        }
      }
    }
  }
}

// ---------------- flash attention: one block per (batch*head, 128-q tile) ----------------
__global__ __launch_bounds__(256)
void flash_attn_kernel(const __bf16* __restrict__ qkv, __bf16* __restrict__ out) {
  __shared__ __align__(16) __bf16 Ks[32][64];      // [kv][dk]
  __shared__ __align__(16) __bf16 Vt[64][32];      // [dk][kv] (transposed)
  __shared__ __align__(16) __bf16 Pb[8][16][32];   // per-wave P relayout buffer

  const int tid = threadIdx.x;
  const int wv  = tid >> 5, lane = tid & 31;
  const int hf  = lane >> 4, l16 = lane & 15;
  const int bh  = blockIdx.x;
  const int b   = bh >> 4, hd = bh & 15;
  const int qt  = blockIdx.y;
  const float scale = 0.125f;  // 1/sqrt(64)

  // Q fragments (2 chunks of K=32 over dk=64) held in registers
  FragBF qf[2];
  {
    const int qrow = qt * 128 + wv * 16 + l16;
    const __bf16* qb = qkv + ((size_t)(b * NSEQ + qrow)) * E3 + hd * DKH;
#pragma unroll
    for (int c = 0; c < 2; ++c) {
      qf[c].q[0] = *(const uint4*)(qb + c * 32 + hf * 8);
      qf[c].q[1] = *(const uint4*)(qb + c * 32 + 16 + hf * 8);
    }
  }

  FragF o[4];
  v8f zero = {};
#pragma unroll
  for (int t = 0; t < 4; ++t) o[t].v = zero;
  float m[8], l[8];
#pragma unroll
  for (int i = 0; i < 8; ++i) { m[i] = -1e30f; l[i] = 0.0f; }

  const int r8 = tid >> 3, c8 = (tid & 7) * 8;

  for (int kt = 0; kt < NSEQ / 32; ++kt) {
    // stage K via async DMA (row-major copy), V via VGPRs (needs transpose)
    const size_t krow = (size_t)(b * NSEQ + kt * 32 + r8) * E3;
    async_b128(lds_addr(&Ks[r8][c8]), qkv + krow + DIMD + hd * DKH + c8);
    union { uint4 u; __bf16 e[8]; } tv;
    tv.u = *(const uint4*)(qkv + krow + 2 * DIMD + hd * DKH + c8);
#pragma unroll
    for (int j = 0; j < 8; ++j) Vt[c8 + j][r8] = tv.e[j];
    wait_asynccnt<0>();
    __syncthreads();

    // S = Q K^T for two 16-wide kv tiles
    FragF s[2];
#pragma unroll
    for (int nt = 0; nt < 2; ++nt) {
      const int kc = nt * 16 + l16;
      FragBF kf;
      s[nt].v = zero;
#pragma unroll
      for (int c = 0; c < 2; ++c) {
        kf.q[0] = *(const uint4*)&Ks[kc][c * 32 + hf * 8];
        kf.q[1] = *(const uint4*)&Ks[kc][c * 32 + 16 + hf * 8];
        s[nt].v = wmma_bf16(qf[c], kf, s[nt].v);
      }
    }

    // online softmax over the 32 new scores per row
#pragma unroll
    for (int i = 0; i < 8; ++i) {
      float s0 = s[0].f[i] * scale;
      float s1 = s[1].f[i] * scale;
      float rm = fmaxf(s0, s1);
#pragma unroll
      for (int off = 1; off < 16; off <<= 1) rm = fmaxf(rm, __shfl_xor(rm, off, 32));
      const float mnew = fmaxf(m[i], rm);
      const float p0 = __expf(s0 - mnew), p1 = __expf(s1 - mnew);
      float rs = p0 + p1;
#pragma unroll
      for (int off = 1; off < 16; off <<= 1) rs += __shfl_xor(rs, off, 32);
      const float alpha = __expf(m[i] - mnew);
      l[i] = l[i] * alpha + rs;
      m[i] = mnew;
#pragma unroll
      for (int t = 0; t < 4; ++t) o[t].f[i] *= alpha;
      const int pr = i + hf * 8;             // C-layout row -> P row
      Pb[wv][pr][l16]      = f2bf(p0);
      Pb[wv][pr][l16 + 16] = f2bf(p1);
    }

    // O += P(16x32) * V(32x64)  (per-wave LDS bounce, DS ops in-order per wave)
    FragBF pf;
    pf.q[0] = *(const uint4*)&Pb[wv][l16][hf * 8];
    pf.q[1] = *(const uint4*)&Pb[wv][l16][16 + hf * 8];
#pragma unroll
    for (int t = 0; t < 4; ++t) {
      const int dc = t * 16 + l16;
      FragBF vf;
      vf.q[0] = *(const uint4*)&Vt[dc][hf * 8];
      vf.q[1] = *(const uint4*)&Vt[dc][16 + hf * 8];
      o[t].v = wmma_bf16(pf, vf, o[t].v);
    }
    __syncthreads();
  }

  // normalize and write [B,N,D] bf16 (heads concatenated)
#pragma unroll
  for (int i = 0; i < 8; ++i) {
    const float inv = 1.0f / l[i];
    const int n = qt * 128 + wv * 16 + i + hf * 8;
    __bf16* ob = out + ((size_t)(b * NSEQ + n)) * DIMD + hd * DKH;
#pragma unroll
    for (int t = 0; t < 4; ++t)
      ob[t * 16 + l16] = f2bf(o[t].f[i] * inv);
  }
}

// ---------------- host-side orchestration ----------------
extern "C" void kernel_launch(void* const* d_in, const int* in_sizes, int n_in,
                              void* d_out, int out_size, void* d_ws, size_t ws_size,
                              hipStream_t stream) {
  const float* x      = (const float*)d_in[0];
  const float* w_qkv  = (const float*)d_in[1];
  const float* w_proj = (const float*)d_in[2];
  const float* b_proj = (const float*)d_in[3];
  const float* w_fc1  = (const float*)d_in[4];
  const float* b_fc1  = (const float*)d_in[5];
  const float* w_fc2  = (const float*)d_in[6];
  const float* b_fc2  = (const float*)d_in[7];
  const float* g1     = (const float*)d_in[8];
  const float* be1    = (const float*)d_in[9];
  const float* g2     = (const float*)d_in[10];
  const float* be2    = (const float*)d_in[11];

  char* ws = (char*)d_ws;
  // workspace layout (bytes)
  __bf16* hb    = (__bf16*)(ws);               // 8192x1024 bf16  (16 MB) LN output (reused for LN2)
  __bf16* w1b   = (__bf16*)(ws + 16777216);    // 3072x1024 bf16  ( 6 MB)
  __bf16* wpb   = (__bf16*)(ws + 23068672);    // 1024x1024 bf16  ( 2 MB)
  __bf16* wf1b  = (__bf16*)(ws + 25165824);    // 4096x1024 bf16  ( 8 MB)
  __bf16* wf2b  = (__bf16*)(ws + 33554432);    // 1024x4096 bf16  ( 8 MB)
  __bf16* qkvb  = (__bf16*)(ws + 41943040);    // 8192x3072 bf16  (50 MB)
  __bf16* attnb = (__bf16*)(ws + 92274688);    // 8192x1024 bf16  (16 MB)
  float*  x1    = (float*) (ws + 109051904);   // 8192x1024 f32   (32 MB)
  __bf16* actb  = (__bf16*)(ws + 142606336);   // 8192x4096 bf16  (64 MB)

  // 1) weights -> bf16 (weights are reused 8192x each; all fit in 192 MB L2)
  cvt_bf16_kernel<<<768, 256, 0, stream>>>(w_qkv,  w1b,  E3 * DIMD);
  cvt_bf16_kernel<<<768, 256, 0, stream>>>(w_proj, wpb,  DIMD * DIMD);
  cvt_bf16_kernel<<<768, 256, 0, stream>>>(w_fc1,  wf1b, HID * DIMD);
  cvt_bf16_kernel<<<768, 256, 0, stream>>>(w_fc2,  wf2b, DIMD * HID);

  // 2) LN1 -> bf16
  ln_bf16_kernel<<<MROWS, 256, 0, stream>>>(x, g1, be1, hb);

  // 3) QKV GEMM (8192x1024x3072) -> bf16
  gemm_bf16_kernel<0><<<dim3(E3 / 128, MROWS / 128), 256, 0, stream>>>(
      hb, w1b, nullptr, nullptr, qkvb, MROWS, E3, DIMD);

  // 4) flash attention
  flash_attn_kernel<<<dim3(BATCH * NHEAD, NSEQ / 128), 256, 0, stream>>>(qkvb, attnb);

  // 5) proj GEMM + bias + residual(x) -> x1 (f32)
  gemm_bf16_kernel<1><<<dim3(DIMD / 128, MROWS / 128), 256, 0, stream>>>(
      attnb, wpb, b_proj, x, x1, MROWS, DIMD, DIMD);

  // 6) LN2 -> bf16 (reuse hb)
  ln_bf16_kernel<<<MROWS, 256, 0, stream>>>(x1, g2, be2, hb);

  // 7) FC1 GEMM + bias + exact GELU -> bf16
  gemm_bf16_kernel<2><<<dim3(HID / 128, MROWS / 128), 256, 0, stream>>>(
      hb, wf1b, b_fc1, nullptr, actb, MROWS, HID, DIMD);

  // 8) FC2 GEMM + bias + residual(x1) -> d_out (f32)
  gemm_bf16_kernel<1><<<dim3(DIMD / 128, MROWS / 128), 256, 0, stream>>>(
      actb, wf2b, b_fc2, x1, (float*)d_out, MROWS, DIMD, HID);
}
